// CEBlock_AP_42356967473847
// MI455X (gfx1250) — matrix operand order repaired and checked by reading, not compiled, not run on hardware
//
#include <hip/hip_runtime.h>
#include <math.h>

// ---------------------------------------------------------------------------
// CEBlock for MI455X (gfx1250): bf16 WMMA GEMMs + fused wave-level attention.
// B=32, N=320 (64 template + 256 search), DIM=768, H=12, dh=64, keep 180/256.
// ~264 GFLOP, working set < 192MB L2 => compute-bound => bf16 WMMA, f32 accum.
// GEMM: 32Mx64N wave tile, register double-buffered K loop, pointer-increment
// addressing, __launch_bounds__(128,1) so nothing spills.
// ---------------------------------------------------------------------------

typedef __attribute__((ext_vector_type(16))) __bf16 v16bf;
typedef __attribute__((ext_vector_type(8)))  __bf16 v8bf;
typedef __attribute__((ext_vector_type(8)))  float  v8f;

#define CB_B    32
#define CB_H    12
#define CB_NT   320
#define CB_DIM  768
#define CB_DH   64
#define CB_LT   64
#define CB_LS   256
#define CB_LK   180
#define CB_NP   244     // 64 + 180 tokens after pruning
#define CB_MQ   (CB_B * CB_NT)   // 10240 rows pre-prune
#define CB_MP   (CB_B * CB_NP)   // 7808 rows post-prune

// ---------------------------------------------------------------------------
// WMMA fragment helpers (layouts per CDNA5 ISA 7.12.2, 16-bit operands)
// ---------------------------------------------------------------------------

__device__ inline v8f wmma_bf16(v16bf a, v16bf b, v8f c) {
  return __builtin_amdgcn_wmma_f32_16x16x32_bf16(
      /*neg_a=*/false, a, /*neg_b=*/false, b,
      /*c_mod=*/(short)0, c, /*reuse_a=*/false, /*reuse_b=*/false);
}

__device__ inline v16bf combine16(v8bf lo, v8bf hi) {
  return __builtin_shufflevector(lo, hi, 0,1,2,3,4,5,6,7,8,9,10,11,12,13,14,15);
}

// A-matrix 16x32 bf16 from row-major [row][k] storage.
// lane<16: m=lane,    elems 0..7 -> K=k0..k0+7,   8..15 -> K=k0+16..k0+23
// lane>=16:m=lane-16, elems 0..7 -> K=k0+8..k0+15,8..15 -> K=k0+24..k0+31
__device__ inline v16bf ld_fragA(const __bf16* __restrict__ base, int ld,
                                 int r0, int k0) {
  int lane = threadIdx.x & 31;
  int m  = lane & 15;
  int kb = (lane & 16) ? 8 : 0;
  const __bf16* p = base + (size_t)(r0 + m) * ld + (k0 + kb);
  return combine16(*(const v8bf*)(p), *(const v8bf*)(p + 16));
}

// B-matrix 32x16 bf16 (K x N). Source is "column-major": mem[n][k] contiguous
// in k (transposed weight / K rows / V^T). lane column n=lane&15,
// same K order as A: lanes<16 K=k0..k0+7,k0+16..k0+23; lanes>=16 K=k0+8..k0+15,k0+24..k0+31.
__device__ inline v16bf ld_fragB(const __bf16* __restrict__ base, int ld,
                                 int n0, int k0) {
  int lane = threadIdx.x & 31;
  int n  = lane & 15;
  int kb = (lane & 16) ? 8 : 0;
  const __bf16* p = base + (size_t)(n0 + n) * ld + (k0 + kb);
  return combine16(*(const v8bf*)(p), *(const v8bf*)(p + 16));
}

// ---------------------------------------------------------------------------
// Weight convert + transpose: W[k][n] f32 -> Wt[n][k] bf16
// ---------------------------------------------------------------------------
__global__ void wtrans_kernel(const float* __restrict__ W,
                              __bf16* __restrict__ Wt, int K, int N) {
  int idx = blockIdx.x * 256 + threadIdx.x;
  if (idx < K * N) {
    int n = idx % N;
    int k = idx / N;
    Wt[(size_t)n * K + k] = (__bf16)W[idx];
  }
}

// ---------------------------------------------------------------------------
// LayerNorm -> bf16 (one 256-thread block per row of DIM=768)
// ---------------------------------------------------------------------------
__global__ void ln_kernel(const float* __restrict__ x,
                          const float* __restrict__ g,
                          const float* __restrict__ be,
                          __bf16* __restrict__ out) {
  __shared__ float red[256];
  int row = blockIdx.x, tid = threadIdx.x;
  const float* xr = x + (size_t)row * CB_DIM;
  float s = 0.f;
  for (int i = tid; i < CB_DIM; i += 256) s += xr[i];
  red[tid] = s; __syncthreads();
  for (int off = 128; off > 0; off >>= 1) {
    if (tid < off) red[tid] += red[tid + off];
    __syncthreads();
  }
  float mu = red[0] / (float)CB_DIM;
  __syncthreads();
  float v = 0.f;
  for (int i = tid; i < CB_DIM; i += 256) { float d = xr[i] - mu; v += d * d; }
  red[tid] = v; __syncthreads();
  for (int off = 128; off > 0; off >>= 1) {
    if (tid < off) red[tid] += red[tid + off];
    __syncthreads();
  }
  float inv = rsqrtf(red[0] / (float)CB_DIM + 1e-5f);
  __bf16* o = out + (size_t)row * CB_DIM;
  for (int i = tid; i < CB_DIM; i += 256)
    o[i] = (__bf16)((xr[i] - mu) * inv * g[i] + be[i]);
}

// ---------------------------------------------------------------------------
// bf16 GEMM: C = A(MxK) * Wt^T + bias. 4 waves/block; wave = 32Mx64N tile,
// 8 accumulators, register double-buffered K loop. Six per-lane fragment
// pointers are computed once and bumped by 64B per chunk -> the loop body is
// pure global_load_b128 + v_add (ptr) + v_wmma; launch_bounds gives the RA
// a full 256-VGPR budget so nothing spills.
// mode 0: scatter into q/k/v buffers (v transposed to [d][n])
// mode 1: + residual -> f32 out. mode 2: GELU -> bf16 out.
// ---------------------------------------------------------------------------
__global__ void __launch_bounds__(128, 1)
gemm_bf16_kernel(const __bf16* __restrict__ A, int lda,
                 const __bf16* __restrict__ Wt, int ldw,
                 const float* __restrict__ bias,
                 int K, int Nfull, int mode,
                 const float* __restrict__ res,
                 float* __restrict__ outf,
                 __bf16* __restrict__ outb,
                 __bf16* __restrict__ qb,
                 __bf16* __restrict__ kb,
                 __bf16* __restrict__ vt) {
  int wave = threadIdx.x >> 5;
  int m0 = blockIdx.x * 32;
  int n0 = blockIdx.y * 256 + wave * 64;

  int lane = threadIdx.x & 31;
  int lsub = lane & 15;
  int kbA = (lane & 16) ? 8 : 0;
  int kbB = (lane & 16) ? 8 : 0;

  // per-lane fragment base pointers (advance by 32 elems = 4 v8bf per chunk)
  const v8bf* pa0 = (const v8bf*)(A  + (size_t)(m0 + lsub)      * lda + kbA);
  const v8bf* pa1 = (const v8bf*)(A  + (size_t)(m0 + 16 + lsub) * lda + kbA);
  const v8bf* pb0 = (const v8bf*)(Wt + (size_t)(n0 + lsub)      * ldw + kbB);
  const v8bf* pb1 = (const v8bf*)(Wt + (size_t)(n0 + 16 + lsub) * ldw + kbB);
  const v8bf* pb2 = (const v8bf*)(Wt + (size_t)(n0 + 32 + lsub) * ldw + kbB);
  const v8bf* pb3 = (const v8bf*)(Wt + (size_t)(n0 + 48 + lsub) * ldw + kbB);

  v8f acc[2][4];
#pragma unroll
  for (int mi = 0; mi < 2; ++mi)
#pragma unroll
    for (int t = 0; t < 4; ++t) acc[mi][t] = (v8f){0,0,0,0,0,0,0,0};

  // preload k=0 fragments (A and B: lo at +0, hi at +16 elems)
  v16bf a0 = combine16(pa0[0], pa0[2]);
  v16bf a1 = combine16(pa1[0], pa1[2]);
  v16bf b0 = combine16(pb0[0], pb0[2]);
  v16bf b1 = combine16(pb1[0], pb1[2]);
  v16bf b2 = combine16(pb2[0], pb2[2]);
  v16bf b3 = combine16(pb3[0], pb3[2]);

  for (int k = 0; k < K; k += 32) {
    v16bf ca0 = a0, ca1 = a1, cb0 = b0, cb1 = b1, cb2 = b2, cb3 = b3;
    pa0 += 4; pa1 += 4; pb0 += 4; pb1 += 4; pb2 += 4; pb3 += 4;
    if (k + 32 < K) {
      __builtin_prefetch(pa0 + 4, 0, 1);
      a0 = combine16(pa0[0], pa0[2]);
      a1 = combine16(pa1[0], pa1[2]);
      b0 = combine16(pb0[0], pb0[2]);
      b1 = combine16(pb1[0], pb1[2]);
      b2 = combine16(pb2[0], pb2[2]);
      b3 = combine16(pb3[0], pb3[2]);
    }
    acc[0][0] = wmma_bf16(ca0, cb0, acc[0][0]);
    acc[0][1] = wmma_bf16(ca0, cb1, acc[0][1]);
    acc[0][2] = wmma_bf16(ca0, cb2, acc[0][2]);
    acc[0][3] = wmma_bf16(ca0, cb3, acc[0][3]);
    acc[1][0] = wmma_bf16(ca1, cb0, acc[1][0]);
    acc[1][1] = wmma_bf16(ca1, cb1, acc[1][1]);
    acc[1][2] = wmma_bf16(ca1, cb2, acc[1][2]);
    acc[1][3] = wmma_bf16(ca1, cb3, acc[1][3]);
  }

  int cn = lsub;
  int rb = (lane & 16) ? 8 : 0;
#pragma unroll
  for (int mi = 0; mi < 2; ++mi) {
#pragma unroll
    for (int t = 0; t < 4; ++t) {
      int n = n0 + 16 * t + cn;
      float bv = bias[n];
#pragma unroll
      for (int r = 0; r < 8; ++r) {
        int m = m0 + mi * 16 + rb + r;
        float v = acc[mi][t][r] + bv;
        if (mode == 0) {
          int b_  = m / CB_NT, tok = m % CB_NT;
          int which = n / CB_DIM, rem = n % CB_DIM;
          int h = rem / CB_DH, d = rem % CB_DH;
          size_t bh = (size_t)(b_ * CB_H + h);
          if (which == 0)      qb[(bh * CB_NT + tok) * CB_DH + d] = (__bf16)v;
          else if (which == 1) kb[(bh * CB_NT + tok) * CB_DH + d] = (__bf16)v;
          else                 vt[(bh * CB_DH + d) * CB_NT + tok] = (__bf16)v;
        } else if (mode == 1) {
          size_t o = (size_t)m * Nfull + n;
          outf[o] = v + res[o];
        } else { // mode 2: exact GELU -> bf16
          float gg = 0.5f * v * (1.f + erff(v * 0.70710678118f));
          outb[(size_t)m * Nfull + n] = (__bf16)gg;
        }
      }
    }
  }
}

// ---------------------------------------------------------------------------
// Fused attention: one wave per (b, h, 16-row q tile).
// scores->LDS, softmax, bf16 P in LDS, P@V via WMMA, o scattered into
// (B*N, DIM) proj-input layout; CE score accumulation via atomics.
// ---------------------------------------------------------------------------
__global__ void __launch_bounds__(32, 1)
attn_kernel(const __bf16* __restrict__ qb,
            const __bf16* __restrict__ kb,
            const __bf16* __restrict__ vt,
            const unsigned char* __restrict__ mask,
            float* __restrict__ sacc,
            __bf16* __restrict__ obuf) {
  __shared__ __align__(16) float  s_tile[16][328];
  __shared__ __align__(16) __bf16 p_tile[16][328];

  int bid = blockIdx.x;
  int qt = bid % (CB_NT / 16);
  int bh = bid / (CB_NT / 16);
  int h  = bh % CB_H;
  int b  = bh / CB_H;
  int lane = threadIdx.x;

  const __bf16* Q  = qb + (size_t)bh * CB_NT * CB_DH;
  const __bf16* Km = kb + (size_t)bh * CB_NT * CB_DH;
  const __bf16* Vt = vt + (size_t)bh * CB_DH * CB_NT;

  v16bf aq0 = ld_fragA(Q, CB_DH, qt * 16, 0);
  v16bf aq1 = ld_fragA(Q, CB_DH, qt * 16, 32);

  int cn = lane & 15;
  int rb = (lane & 16) ? 8 : 0;

  // S = Q K^T * dh^-0.5
  for (int j = 0; j < CB_NT / 16; ++j) {
    v8f c = (v8f){0,0,0,0,0,0,0,0};
    c = wmma_bf16(aq0, ld_fragB(Km, CB_DH, j * 16, 0),  c);
    c = wmma_bf16(aq1, ld_fragB(Km, CB_DH, j * 16, 32), c);
    int nn = j * 16 + cn;
#pragma unroll
    for (int r = 0; r < 8; ++r) s_tile[rb + r][nn] = c[r] * 0.125f;
  }
  __syncthreads();

  // softmax over 320 cols; 2 lanes per row
  int row = lane & 15;
  int half = lane >> 4;
  float mx = -1e30f;
  for (int c2 = half; c2 < CB_NT; c2 += 2) mx = fmaxf(mx, s_tile[row][c2]);
  mx = fmaxf(mx, __shfl_xor(mx, 16, 32));
  float sum = 0.f;
  for (int c2 = half; c2 < CB_NT; c2 += 2) {
    float e = __expf(s_tile[row][c2] - mx);
    s_tile[row][c2] = e;
    sum += e;
  }
  sum += __shfl_xor(sum, 16, 32);
  float inv = 1.f / sum;
  for (int c2 = half; c2 < CB_NT; c2 += 2)
    p_tile[row][c2] = (__bf16)(s_tile[row][c2] * inv);
  __syncthreads();

  // O = P @ V  (16 x 64)
  v8f co[4];
#pragma unroll
  for (int t = 0; t < 4; ++t) co[t] = (v8f){0,0,0,0,0,0,0,0};
  for (int kc = 0; kc < CB_NT / 32; ++kc) {
    v16bf ap = ld_fragA(&p_tile[0][0], 328, 0, kc * 32);
#pragma unroll
    for (int t = 0; t < 4; ++t)
      co[t] = wmma_bf16(ap, ld_fragB(Vt, CB_NT, t * 16, kc * 32), co[t]);
  }
  // scatter o into (B*N, DIM) layout at col h*64
#pragma unroll
  for (int t = 0; t < 4; ++t) {
    int d = t * 16 + cn;
#pragma unroll
    for (int r = 0; r < 8; ++r) {
      int tok = qt * 16 + rb + r;
      obuf[((size_t)(b * CB_NT + tok)) * CB_DIM + h * CB_DH + d] = (__bf16)co[t][r];
    }
  }

  // CE score accumulation: rows in template region only (qt < 4)
  if (qt < 4) {
    float w[16];
#pragma unroll
    for (int r = 0; r < 16; ++r)
      w[r] = mask[b * CB_LT + qt * 16 + r] ? 1.f : 0.f;
    for (int c2 = lane; c2 < CB_LS; c2 += 32) {
      float s2 = 0.f;
#pragma unroll
      for (int r = 0; r < 16; ++r)
        s2 += w[r] * (float)p_tile[r][CB_LT + c2];
      atomicAdd(&sacc[b * CB_LS + c2], s2);
    }
  }
}

// ---------------------------------------------------------------------------
// Zero-fill helper (sacc)
// ---------------------------------------------------------------------------
__global__ void zero_kernel(float* __restrict__ p, int n) {
  int i = blockIdx.x * 256 + threadIdx.x;
  if (i < n) p[i] = 0.f;
}

// ---------------------------------------------------------------------------
// Per-batch descending argsort (bitonic, index tie-break = stable argsort),
// emits keep/rem indices (as float, via gsi) and the order for gathering.
// ---------------------------------------------------------------------------
__global__ void sort_kernel(const float* __restrict__ sacc,
                            const unsigned char* __restrict__ mask,
                            const int* __restrict__ gsi,
                            int* __restrict__ order_out,
                            float* __restrict__ keep_out,
                            float* __restrict__ rem_out) {
  __shared__ float sv[CB_LS];
  __shared__ int   si[CB_LS];
  __shared__ float cntsh;
  int b = blockIdx.x, tid = threadIdx.x;
  if (tid == 0) {
    float c = 0.f;
    for (int t = 0; t < CB_LT; ++t) c += mask[b * CB_LT + t] ? 1.f : 0.f;
    cntsh = c;
  }
  __syncthreads();
  sv[tid] = sacc[b * CB_LS + tid] / (cntsh * (float)CB_H);
  si[tid] = tid;
  __syncthreads();
  for (int k = 2; k <= CB_LS; k <<= 1) {
    for (int j = k >> 1; j > 0; j >>= 1) {
      int ixj = tid ^ j;
      if (ixj > tid) {
        float v1 = sv[tid], v2 = sv[ixj];
        int i1 = si[tid], i2 = si[ixj];
        bool p2first = (v2 > v1) || (v2 == v1 && i2 < i1); // ixj should precede
        bool p1first = (v1 > v2) || (v1 == v2 && i1 < i2);
        bool descBlock = ((tid & k) == 0);
        if (descBlock ? p2first : p1first) {
          sv[tid] = v2; sv[ixj] = v1;
          si[tid] = i2; si[ixj] = i1;
        }
      }
      __syncthreads();
    }
  }
  int o = si[tid];
  order_out[b * CB_LS + tid] = o;
  float gv = (float)gsi[b * CB_LS + o];
  if (tid < CB_LK) keep_out[b * CB_LK + tid] = gv;
  else             rem_out[b * (CB_LS - CB_LK) + (tid - CB_LK)] = gv;
}

// ---------------------------------------------------------------------------
// Gather pruned token rows: xp (B,244,768) from xa (B,320,768)
// ---------------------------------------------------------------------------
__global__ void gather_kernel(const float* __restrict__ xa,
                              const int* __restrict__ order,
                              float* __restrict__ xp) {
  int row = blockIdx.x;            // 0 .. B*NP-1
  int b = row / CB_NP, i = row % CB_NP;
  int src = (i < CB_LT) ? i : (CB_LT + order[b * CB_LS + (i - CB_LT)]);
  const float* s = xa + ((size_t)b * CB_NT + src) * CB_DIM;
  float* d = xp + (size_t)row * CB_DIM;
  for (int c = threadIdx.x; c < CB_DIM; c += 256) d[c] = s[c];
}

__global__ void i2f_kernel(const int* __restrict__ in, float* __restrict__ out,
                           int n) {
  int i = blockIdx.x * 256 + threadIdx.x;
  if (i < n) out[i] = (float)in[i];
}

// ---------------------------------------------------------------------------
// Host launch
// ---------------------------------------------------------------------------
extern "C" void kernel_launch(void* const* d_in, const int* in_sizes, int n_in,
                              void* d_out, int out_size, void* d_ws, size_t ws_size,
                              hipStream_t stream) {
  (void)in_sizes; (void)n_in; (void)out_size; (void)ws_size;

  const float* x_in[2]  = {(const float*)d_in[0], (const float*)d_in[1]};
  const int*   git      = (const int*)d_in[2];
  const int*   gsi[2]   = {(const int*)d_in[3], (const int*)d_in[4]};
  const unsigned char* cmask = (const unsigned char*)d_in[5];
  const float* g1    = (const float*)d_in[6];
  const float* b1    = (const float*)d_in[7];
  const float* Wqkv  = (const float*)d_in[8];
  const float* bqkv  = (const float*)d_in[9];
  const float* Wproj = (const float*)d_in[10];
  const float* bproj = (const float*)d_in[11];
  const float* g2    = (const float*)d_in[12];
  const float* b2    = (const float*)d_in[13];
  const float* W1    = (const float*)d_in[14];
  const float* bm1   = (const float*)d_in[15];
  const float* W2    = (const float*)d_in[16];
  const float* bm2   = (const float*)d_in[17];

  float* out = (float*)d_out;
  const size_t XSZ = (size_t)CB_B * CB_NP * CB_DIM;
  float* out_x[2] = {out, out + XSZ};
  float* out_git  = out + 2 * XSZ;
  float* out_k[2] = {out_git + CB_B * CB_LT,
                     out_git + CB_B * CB_LT + CB_B * CB_LK};
  float* out_r[2] = {out_k[1] + CB_B * CB_LK,
                     out_k[1] + CB_B * CB_LK + CB_B * (CB_LS - CB_LK)};

  // scratch carve-out
  char* wp = (char*)d_ws;
  auto alloc = [&](size_t bytes) -> void* {
    void* r = (void*)wp;
    wp += (bytes + 255) & ~(size_t)255;
    return r;
  };
  __bf16* Wqkvt  = (__bf16*)alloc((size_t)3 * CB_DIM * CB_DIM * 2);
  __bf16* Wprojt = (__bf16*)alloc((size_t)CB_DIM * CB_DIM * 2);
  __bf16* W1t    = (__bf16*)alloc((size_t)4 * CB_DIM * CB_DIM * 2);
  __bf16* W2t    = (__bf16*)alloc((size_t)4 * CB_DIM * CB_DIM * 2);
  __bf16* xln    = (__bf16*)alloc((size_t)CB_MQ * CB_DIM * 2);
  __bf16* qb     = (__bf16*)alloc((size_t)CB_B * CB_H * CB_NT * CB_DH * 2);
  __bf16* kb     = (__bf16*)alloc((size_t)CB_B * CB_H * CB_NT * CB_DH * 2);
  __bf16* vt     = (__bf16*)alloc((size_t)CB_B * CB_H * CB_NT * CB_DH * 2);
  __bf16* obuf   = (__bf16*)alloc((size_t)CB_MQ * CB_DIM * 2);
  float*  xa     = (float*) alloc((size_t)CB_MQ * CB_DIM * 4);
  float*  sacc   = (float*) alloc((size_t)CB_B * CB_LS * 4);
  int*    order  = (int*)   alloc((size_t)CB_B * CB_LS * 4);
  float*  xp     = (float*) alloc((size_t)CB_MP * CB_DIM * 4);
  __bf16* xln2   = (__bf16*)alloc((size_t)CB_MP * CB_DIM * 2);
  __bf16* hbuf   = (__bf16*)alloc((size_t)CB_MP * 4 * CB_DIM * 2);

  dim3 tb256(256), tb128(128), tb32(32);

  // weights -> transposed bf16 (once per call; deterministic)
  wtrans_kernel<<<dim3((CB_DIM*3*CB_DIM+255)/256), tb256, 0, stream>>>(Wqkv,  Wqkvt,  CB_DIM, 3*CB_DIM);
  wtrans_kernel<<<dim3((CB_DIM*CB_DIM+255)/256),   tb256, 0, stream>>>(Wproj, Wprojt, CB_DIM, CB_DIM);
  wtrans_kernel<<<dim3((CB_DIM*4*CB_DIM+255)/256), tb256, 0, stream>>>(W1,    W1t,    CB_DIM, 4*CB_DIM);
  wtrans_kernel<<<dim3((4*CB_DIM*CB_DIM+255)/256), tb256, 0, stream>>>(W2,    W2t,    4*CB_DIM, CB_DIM);

  for (int br = 0; br < 2; ++br) {
    // LN1 -> bf16
    ln_kernel<<<dim3(CB_MQ), tb256, 0, stream>>>(x_in[br], g1, b1, xln);

    // QKV GEMM: (10240x768) @ (768x2304), scatter into q/k/v^T
    gemm_bf16_kernel<<<dim3(CB_MQ/32, (3*CB_DIM)/256), tb128, 0, stream>>>(
        xln, CB_DIM, Wqkvt, CB_DIM, bqkv, CB_DIM, 3*CB_DIM, /*mode=*/0,
        nullptr, nullptr, nullptr, qb, kb, vt);

    zero_kernel<<<dim3((CB_B*CB_LS+255)/256), tb256, 0, stream>>>(sacc, CB_B*CB_LS);

    // fused attention, one wave per (b,h,qtile)
    attn_kernel<<<dim3(CB_B*CB_H*(CB_NT/16)), tb32, 0, stream>>>(
        qb, kb, vt, cmask, sacc, obuf);

    // proj GEMM + residual -> xa (f32)
    gemm_bf16_kernel<<<dim3(CB_MQ/32, CB_DIM/256), tb128, 0, stream>>>(
        obuf, CB_DIM, Wprojt, CB_DIM, bproj, CB_DIM, CB_DIM, /*mode=*/1,
        x_in[br], xa, nullptr, nullptr, nullptr, nullptr);

    // CE: normalize + argsort + emit keep/rem
    sort_kernel<<<dim3(CB_B), tb256, 0, stream>>>(
        sacc, cmask, gsi[br], order, out_k[br], out_r[br]);

    // gather pruned tokens -> xp
    gather_kernel<<<dim3(CB_MP), tb256, 0, stream>>>(xa, order, xp);

    // LN2 -> bf16
    ln_kernel<<<dim3(CB_MP), tb256, 0, stream>>>(xp, g2, b2, xln2);

    // MLP1: (7808x768) @ (768x3072), GELU -> bf16
    gemm_bf16_kernel<<<dim3(CB_MP/32, (4*CB_DIM)/256), tb128, 0, stream>>>(
        xln2, CB_DIM, W1t, CB_DIM, bm1, CB_DIM, 4*CB_DIM, /*mode=*/2,
        nullptr, nullptr, hbuf, nullptr, nullptr, nullptr);

    // MLP2: (7808x3072) @ (3072x768) + residual -> d_out x region
    gemm_bf16_kernel<<<dim3(CB_MP/32, CB_DIM/256), tb128, 0, stream>>>(
        hbuf, 4*CB_DIM, W2t, 4*CB_DIM, bm2, 4*CB_DIM, CB_DIM, /*mode=*/1,
        xp, out_x[br], nullptr, nullptr, nullptr, nullptr);
  }

  // global_index_template passthrough (as float)
  i2f_kernel<<<dim3((CB_B*CB_LT+255)/256), tb256, 0, stream>>>(
      git, out_git, CB_B*CB_LT);
}
